// BottomUpInferenceLayer_21148418966132
// MI455X (gfx1250) — matrix operand order repaired and checked by reading, not compiled, not run on hardware
//
#include <hip/hip_runtime.h>
#include <hip/hip_bf16.h>

// ---------------- problem constants (match reference) ----------------
#define B_   16
#define H_   512
#define W_   512
#define NN   15      // N_NODES
#define NE   14      // N_EDGES
#define KTOP 32
#define PSAMP 10
#define THR_ 0.2f
#define HC   256     // cm output H=W (stride 2)
#define WC   256
#define HP   128     // paf output H=W (stride 4)
#define WP   128

typedef __attribute__((ext_vector_type(16))) _Float16 v16h;
typedef __attribute__((ext_vector_type(8)))  float    v8f;

__device__ __forceinline__ bool better(float v1, int i1, float v2, int i2) {
    // lax.top_k ordering: value descending, ties broken by lower index
    return (v1 > v2) || (v1 == v2 && i1 < i2);
}

__device__ __forceinline__ float sgnf(float v) {
    return (v > 0.f) ? 1.f : ((v < 0.f) ? -1.f : 0.f);
}

// ---------------------------------------------------------------------
// Conv as im2col GEMM on the matrix pipe:
//   one wave => 16 consecutive output pixels (M) x 16 output channels (N),
//   K = 32 (9 real 3x3 taps, rest zero), V_WMMA_F32_16X16X32_F16.
// Layouts per CDNA5 ISA 7.12.2:
//   A (16x32 f16): lane m=lane&15; elem e -> K = (lane<16?0:8) + e + (e>=8?8:0)
//   B (32x16 f16): lane n=lane&15; elem e -> K = (lane<16?0:8) + e + (e>=8?8:0)
//   C (16x16 f32): vgpr r -> M = r + (lane<16?0:8), N = lane&15
// ---------------------------------------------------------------------
template <int STRIDE, int COUT, int OUTDIM, int NCT, bool SIG>
__global__ __launch_bounds__(256) void conv_wmma_kernel(
    const float* __restrict__ img, const float* __restrict__ wt,
    const float* __restrict__ bias, float* __restrict__ out) {
    constexpr int WTILES = OUTDIM / 16;
    const int lane = threadIdx.x & 31;
    const int wave = blockIdx.x * (blockDim.x >> 5) + (threadIdx.x >> 5);
    const int tilesPerImg = OUTDIM * WTILES;
    const int b   = wave / tilesPerImg;
    const int rem = wave % tilesPerImg;
    const int y   = rem / WTILES;
    const int x0  = (rem % WTILES) * 16;

    const int m      = lane & 15;
    const int kbaseA = (lane < 16) ? 0 : 8;
    const int iy0    = y * STRIDE;             // SAME pad_lo == 0 for both convs
    const int ix0    = (x0 + m) * STRIDE;

    __builtin_prefetch(&img[(size_t)(b * H_ + iy0) * W_ + ix0], 0, 0);

    v16h a;
#pragma unroll
    for (int e = 0; e < 16; ++e) {
        const int k = kbaseA + e + ((e >= 8) ? 8 : 0);
        float v = 0.f;
        if (k < 9) {
            const int ky = k / 3, kx = k % 3;
            const int iy = iy0 + ky, ix = ix0 + kx;
            if (iy < H_ && ix < W_) v = img[(size_t)(b * H_ + iy) * W_ + ix];
        }
        a[e] = (_Float16)v;
    }

    const int n      = lane & 15;
    const int kbaseB = (lane < 16) ? 0 : 8;

#pragma unroll
    for (int ct = 0; ct < NCT; ++ct) {
        const int ch = ct * 16 + n;
        v16h bm;
#pragma unroll
        for (int e = 0; e < 16; ++e) {
            const int k = kbaseB + e + ((e >= 8) ? 8 : 0);
            float v = 0.f;
            if (k < 9 && ch < COUT) v = wt[k * COUT + ch];  // HWIO, I==1
            bm[e] = (_Float16)v;
        }
        v8f c = {};
        c = __builtin_amdgcn_wmma_f32_16x16x32_f16(
            /*neg_a=*/false, a, /*neg_b=*/false, bm,
            /*c_mod=*/(short)0, c, /*reuse_a=*/false, /*reuse_b=*/false);
        if (ch < COUT) {
            const float bv = bias[ch];
#pragma unroll
            for (int r = 0; r < 8; ++r) {
                const int M = r + ((lane < 16) ? 0 : 8);
                float v = c[r] + bv;
                if (SIG) v = 1.f / (1.f + __expf(-v));
                out[((size_t)(b * OUTDIM + y) * OUTDIM + (x0 + M)) * COUT + ch] = v;
            }
        }
    }
}

// ---------------------------------------------------------------------
// Fused 3x3 NMS + exact top-32 per (b,node), one workgroup each.
// 128 threads x contiguous 512-element slices -> per-thread sorted top-32
// in LDS (34 KB), then 32-round k-way merge reproducing lax.top_k order.
// Threads 0..31 then do sub-pixel refinement and emit peaks / peak_vals.
// ---------------------------------------------------------------------
__global__ __launch_bounds__(128) void peak_topk_kernel(
    const float* __restrict__ cms, float* __restrict__ out_peaks,
    float* __restrict__ out_vals) {
    __shared__ float cv[128 * 32];
    __shared__ int   ci[128 * 32];
    __shared__ int   heads[128];
    __shared__ float rv[128];
    __shared__ int   ri[128];
    __shared__ int   rt[128];
    __shared__ float sv[KTOP];
    __shared__ int   si[KTOP];

    const int bn   = blockIdx.x;
    const int b    = bn / NN;
    const int node = bn % NN;
    const int t    = threadIdx.x;
    const float NEG = -__builtin_huge_valf();

    float* mv = &cv[t * 32];
    int*   mi = &ci[t * 32];
    for (int j = 0; j < 32; ++j) { mv[j] = NEG; mi[j] = 0x7fffffff; }

    const size_t mapBase = (size_t)b * HC * WC * NN + node;
    const int slice = (HC * WC) / 128;  // 512
    const int base  = t * slice;
    for (int s = 0; s < slice; ++s) {
        const int idx = base + s;
        const int r = idx >> 8;        // WC == 256
        const int c = idx & (WC - 1);
        const float v = cms[mapBase + (size_t)idx * NN];
        float mx = NEG;                // reduce_window SAME pads with -inf
        for (int dy = -1; dy <= 1; ++dy) {
            const int rr = r + dy;
            if (rr < 0 || rr >= HC) continue;
            for (int dx = -1; dx <= 1; ++dx) {
                const int cc = c + dx;
                if (cc < 0 || cc >= WC) continue;
                mx = fmaxf(mx, cms[mapBase + ((size_t)rr * WC + cc) * NN]);
            }
        }
        const bool  peak = (v == mx) && (v > THR_);
        const float cand = peak ? v : NEG;
        if (better(cand, idx, mv[31], mi[31])) {
            int j = 31;
            while (j > 0 && better(cand, idx, mv[j - 1], mi[j - 1])) {
                mv[j] = mv[j - 1]; mi[j] = mi[j - 1]; --j;
            }
            mv[j] = cand; mi[j] = idx;
        }
    }
    heads[t] = 0;
    __syncthreads();

    for (int kk = 0; kk < KTOP; ++kk) {
        const int h = heads[t];
        rv[t] = (h < 32) ? cv[t * 32 + h] : NEG;
        ri[t] = (h < 32) ? ci[t * 32 + h] : 0x7fffffff;
        rt[t] = t;
        __syncthreads();
        for (int off = 64; off > 0; off >>= 1) {
            if (t < off) {
                if (better(rv[t + off], ri[t + off], rv[t], ri[t])) {
                    rv[t] = rv[t + off]; ri[t] = ri[t + off]; rt[t] = rt[t + off];
                }
            }
            __syncthreads();
        }
        if (t == 0) { sv[kk] = rv[0]; si[kk] = ri[0]; heads[rt[0]]++; }
        __syncthreads();
    }

    if (t < KTOP) {
        const float tv  = sv[t];
        const int   idx = si[t];
        const int   r   = idx >> 8;
        const int   c   = idx & (WC - 1);
        auto G = [&](int rr, int cc) -> float {
            rr = rr < 0 ? 0 : (rr > HC - 1 ? HC - 1 : rr);
            cc = cc < 0 ? 0 : (cc > WC - 1 ? WC - 1 : cc);
            return cms[mapBase + ((size_t)rr * WC + cc) * NN];
        };
        const float dxv = 0.25f * sgnf(G(r, c + 1) - G(r, c - 1));
        const float dyv = 0.25f * sgnf(G(r + 1, c) - G(r - 1, c));
        const size_t p  = (size_t)(b * NN + node) * KTOP + t;
        out_peaks[p * 2 + 0] = ((float)c + dxv) * 2.0f;  // CM_STRIDE
        out_peaks[p * 2 + 1] = ((float)r + dyv) * 2.0f;
        out_vals[p]          = (tv > THR_) ? tv : 0.0f;
    }
}

// ---------------------------------------------------------------------
// Line-integral edge scores: one block per (b,edge), 1024 peak pairs,
// 10 PAF samples each. pafs are L2-resident (29 MB << 192 MB L2).
// ---------------------------------------------------------------------
__global__ __launch_bounds__(256) void line_score_kernel(
    const float* __restrict__ pafs, const float* __restrict__ peaks,
    const float* __restrict__ pvals, const int* __restrict__ edge_inds,
    float* __restrict__ out_scores) {
    const int be = blockIdx.x;
    const int b  = be / NE;
    const int e  = be % NE;
    const int e0 = edge_inds[e * 2 + 0];
    const int e1 = edge_inds[e * 2 + 1];

    for (int pair = threadIdx.x; pair < KTOP * KTOP; pair += blockDim.x) {
        const int s  = pair >> 5;
        const int tt = pair & 31;
        const size_t sp = ((size_t)(b * NN + e0) * KTOP + s) * 2;
        const size_t dp = ((size_t)(b * NN + e1) * KTOP + tt) * 2;
        const float sx = peaks[sp + 0], sy = peaks[sp + 1];
        const float ddx = peaks[dp + 0] - sx;
        const float ddy = peaks[dp + 1] - sy;
        const float inv = 1.0f / sqrtf(ddx * ddx + ddy * ddy + 1e-8f);
        const float ux = ddx * inv, uy = ddy * inv;

        float acc = 0.f;
#pragma unroll
        for (int p = 0; p < PSAMP; ++p) {
            const float tp = (float)p * (1.0f / (PSAMP - 1));  // linspace(0,1,10)
            const float px = sx + tp * ddx;
            const float py = sy + tp * ddy;
            int xi = (int)rintf(px * 0.25f);  // / PAF_STRIDE, round-to-even
            int yi = (int)rintf(py * 0.25f);
            xi = xi < 0 ? 0 : (xi > WP - 1 ? WP - 1 : xi);
            yi = yi < 0 ? 0 : (yi > HP - 1 ? HP - 1 : yi);
            const float* pf =
                &pafs[(((size_t)b * HP + yi) * WP + xi) * (2 * NE) + e * 2];
            acc += pf[0] * ux + pf[1] * uy;
        }
        const bool valid = (pvals[(size_t)(b * NN + e0) * KTOP + s] > THR_) &&
                           (pvals[(size_t)(b * NN + e1) * KTOP + tt] > THR_);
        out_scores[((size_t)(b * NE + e) * KTOP + s) * KTOP + tt] =
            valid ? acc * (1.0f / PSAMP) : -1.0f;
    }
}

// ---------------------------------------------------------------------
extern "C" void kernel_launch(void* const* d_in, const int* in_sizes, int n_in,
                              void* d_out, int out_size, void* d_ws,
                              size_t ws_size, hipStream_t stream) {
    const float* images = (const float*)d_in[0];
    const float* cmk    = (const float*)d_in[1];
    const float* cmb    = (const float*)d_in[2];
    const float* pafk   = (const float*)d_in[3];
    const float* pafb   = (const float*)d_in[4];
    const int*   einds  = (const int*)d_in[5];

    float* out     = (float*)d_out;
    float* peaks   = out;                              // B*NN*K*2 = 15360
    float* pvals   = out + (size_t)B_ * NN * KTOP * 2; // B*NN*K   = 7680
    float* lscores = pvals + (size_t)B_ * NN * KTOP;   // B*NE*K*K = 229376

    float* cms  = (float*)d_ws;                              // 62.9 MB
    float* pafs = cms + (size_t)B_ * HC * WC * NN;           // 29.4 MB

    // cm conv: 16*256*(256/16)=65536 waves -> 8192 blocks of 8 waves (exact)
    conv_wmma_kernel<2, NN, HC, 1, true>
        <<<8192, 256, 0, stream>>>(images, cmk, cmb, cms);
    // paf conv: 16*128*(128/16)=16384 waves -> 2048 blocks (exact), 2 ch-tiles
    conv_wmma_kernel<4, 2 * NE, HP, 2, false>
        <<<2048, 256, 0, stream>>>(images, pafk, pafb, pafs);
    // exact top-32 per (b,node): 240 workgroups
    peak_topk_kernel<<<B_ * NN, 128, 0, stream>>>(cms, peaks, pvals);
    // edge scores: 224 workgroups
    line_score_kernel<<<B_ * NE, 256, 0, stream>>>(pafs, peaks, pvals, einds,
                                                   lscores);
}
